// TransformerModelLarge_29326036697777
// MI455X (gfx1250) — matrix operand order repaired and checked by reading, not compiled, not run on hardware
//
#include <hip/hip_runtime.h>

// ---------------------------------------------------------------------------
// Fully-fused 16-layer network for MI455X (gfx1250).
// - 8 waves/WG (2 per SIMD32): each wave-pair shares a 16-row M-tile, split by
//   N-tile (jSel) inside each 32-row weight slab.
// - bf16 WMMA (v_wmma_f32_16x16x32_bf16), f32 accumulate, DUAL accumulator
//   chains + sched_group_barrier [DS4|WMMA2] pattern for 2-deep B pipelining.
// - Weight slabs staged via Tensor Data Mover (tensor_load_to_lds, TENSORcnt),
//   triple-buffered -> single barrier per stage.
// - A-panel (16 k-tiles) cached in 128 VGPRs per layer for the 512-K layers.
// ---------------------------------------------------------------------------

typedef unsigned int       u32;
typedef unsigned short     u16;
typedef unsigned long long u64;
typedef __attribute__((ext_vector_type(16))) __bf16 bf16x16;
typedef __attribute__((ext_vector_type(8)))  float  f32x8;
typedef __attribute__((ext_vector_type(4)))  u32    u32x4;
typedef __attribute__((ext_vector_type(8)))  int    i32x8;
typedef __attribute__((ext_vector_type(4)))  int    i32x4;

#define DIM          512
#define IN_DIM       2048
#define BATCH        32768
#define LDS_STRIDE   520   // activation row stride (bf16): 260 dw % 64 banks = 4
#define SLAB_STRIDE  520   // slab row stride: 16 B-readers land on banks 4n..4n+3
#define SLAB_ROWS    32    // output-rows per slab chunk (2 WMMA N-tiles)
#define ROWS_PER_WG  64    // 4 M-tiles x 16 rows
#define THREADS      256   // 8 waves: wave&3 = M-tile, wave>>2 = N-tile (jSel)
#define ACT_BYTES    (3 * ROWS_PER_WG * LDS_STRIDE * 2)
#define SLAB_BYTES   (SLAB_ROWS * SLAB_STRIDE * 2)
#define SMEM_BYTES   (ACT_BYTES + 3 * SLAB_BYTES)   // 199,680 + 99,840 = 299,520 B
#define SLAB0_OFF    ((u32)ACT_BYTES)
#define SLAB1_OFF    ((u32)(ACT_BYTES + SLAB_BYTES))
#define SLAB2_OFF    ((u32)(ACT_BYTES + 2 * SLAB_BYTES))

#if __has_builtin(__builtin_amdgcn_tensor_load_to_lds)
#define HAS_TDM 1
#if __has_include(<hip/amd_detail/amd_gfx1250_TDM.h>)
#define TDM_SIX_ARGS 1
#endif
#else
#define HAS_TDM 0
#endif

// Per-iteration scheduling intent: 4 DS reads (next B pairs) then 2 WMMAs.
// Forces both B double-buffers live simultaneously -> 2-deep LDS pipelining.
#if __has_builtin(__builtin_amdgcn_sched_group_barrier)
#define SCHED_DS4_WMMA2()                                  \
  do {                                                     \
    __builtin_amdgcn_sched_group_barrier(0x100, 4, 0);     \
    __builtin_amdgcn_sched_group_barrier(0x008, 2, 0);     \
  } while (0)
#else
#define SCHED_DS4_WMMA2() do {} while (0)
#endif

enum Act { ACT_NONE = 0, ACT_RELU = 1, ACT_RELU_RES = 2 };

__device__ __forceinline__ u16 f2bf(float f) {
  u32 u = __float_as_uint(f);
  return (u16)((u + 0x7fffu + ((u >> 16) & 1u)) >> 16);   // RNE
}
__device__ __forceinline__ float bf2f(u16 h) { return __uint_as_float(((u32)h) << 16); }

union ABu { uint4 q[2]; bf16x16 v; };

// A-matrix 16x32 bf16 (ISA 7.12.2): p points at (row, half*8); chunks +0, +16 elems.
__device__ __forceinline__ bf16x16 load_a(const u16* p) {
  ABu u;
  u.q[0] = *(const uint4*)(p);
  u.q[1] = *(const uint4*)(p + 16);
  return u.v;
}
// B-matrix 32x16 bf16: lane = column, p points at half*8; chunks +0, +16 elems.
__device__ __forceinline__ bf16x16 load_b(const u16* p) {
  ABu u;
  u.q[0] = *(const uint4*)(p);
  u.q[1] = *(const uint4*)(p + 16);
  return u.v;
}

#if HAS_TDM
__device__ __forceinline__ void tdm_wait(int pending) {
#if __has_builtin(__builtin_amdgcn_s_wait_tensorcnt)
  if (pending) __builtin_amdgcn_s_wait_tensorcnt(1);
  else         __builtin_amdgcn_s_wait_tensorcnt(0);
#else
  if (pending) asm volatile("s_wait_tensorcnt 0x1" ::: "memory");
  else         asm volatile("s_wait_tensorcnt 0x0" ::: "memory");
#endif
}

// Issue one TDM load: 2D tile [SLAB_ROWS x 512] bf16 from row-major W
// (row stride kFull elems) into LDS at ldsOff with 520-elem padded rows
// (pad_interval=256 dwords -> every 512 bf16, pad_amount=4 dwords -> 8 bf16).
__device__ __forceinline__ void tdm_load_slab(u32 ldsOff, const u16* gsrc, u32 kFull) {
  const u64 ga = (u64)(size_t)gsrc;
  u32x4 g0;
  g0.x = 1u;                                            // count=1 valid user D#
  g0.y = ldsOff;                                        // lds_addr (bytes)
  g0.z = (u32)ga;                                       // global_addr[31:0]
  g0.w = ((u32)(ga >> 32) & 0x01ffffffu) | 0x80000000u; // addr[56:32] | type=2
  i32x8 g1;
  g1[0] = (int)((1u << 16) | (1u << 20) | (7u << 22) | (3u << 25));
         // data_size=2B | pad_enable | pad_interval=256dw | pad_amount=4dw
  g1[1] = (int)(512u << 16);                            // tensor_dim0=512 (lo16 @bit48)
  g1[2] = (int)((u32)SLAB_ROWS << 16);                  // dim0 hi=0 | tensor_dim1=32 lo16
  g1[3] = (int)(512u << 16);                            // dim1 hi=0 | tile_dim0=512
  g1[4] = (int)SLAB_ROWS;                               // tile_dim1=32, tile_dim2=0
  g1[5] = (int)kFull;                                   // tensor_dim0_stride lo32
  g1[6] = 0;                                            // stride hi | dim1_stride lo
  g1[7] = 0;
  const i32x4 z4 = {0, 0, 0, 0};
#if defined(TDM_SIX_ARGS)
  const i32x8 z8 = {0, 0, 0, 0, 0, 0, 0, 0};
  __builtin_amdgcn_tensor_load_to_lds(g0, g1, z4, z4, z8, 0);
#else
  __builtin_amdgcn_tensor_load_to_lds(g0, g1, z4, z4, 0);
#endif
}
#else
// Fallback: synchronous cooperative copy of one slab chunk (256 threads).
__device__ __forceinline__ void copy_slab(u16* slab, const u16* Wsrc, int kFull) {
  const int t = threadIdx.x;
  const int r = t >> 3, sub = t & 7;                    // 8 threads per 512-col row
  const uint4* src = (const uint4*)(Wsrc + (size_t)r * kFull + sub * 64);
  uint4* dst = (uint4*)(slab + r * SLAB_STRIDE + sub * 64);
#pragma unroll
  for (int i = 0; i < 8; ++i) dst[i] = src[i];
}
#endif

// One linear layer; each wave computes its 16-row M-tile x its jSel N-tile of
// every 32-col chunk. A_GLOBAL: A from global bf16 (aBase = lane row ptr),
// KB sub-slabs of 512 K. Else A from LDS, panel cached in 128 VGPRs.
template <bool A_GLOBAL, int KB>
__device__ void gemm_tiled(const u16* __restrict__ aBase, u16* __restrict__ dstW,
                           const u16* __restrict__ resW, const u16* __restrict__ W,
                           const float* __restrict__ bias,
                           u16* slab0, u16* slab1, u16* slab2,
                           int act, int ln16, int half, int jSel, bool issuer) {
  const int Kfull = A_GLOBAL ? IN_DIM : DIM;
  const int S = 16 * KB;                     // 16 col-chunks x KB K-slabs
  const int bRow = (jSel * 16 + ln16) * SLAB_STRIDE;
  (void)issuer; (void)slab1; (void)slab2;

  __syncthreads();                 // prev layer epilogues (both pair waves) retired
#if HAS_TDM
  if (issuer) tdm_load_slab(SLAB0_OFF, W, (u32)Kfull);           // stage 0
  const u16* slabCur = slab0;
  u32 offNext = SLAB1_OFF;
#endif
  bf16x16 Atile[16];                         // per-layer A panel cache (LDS path)
  if constexpr (!A_GLOBAL) {
#pragma unroll
    for (int kt = 0; kt < 16; ++kt)
      Atile[kt] = load_a(aBase + ln16 * LDS_STRIDE + kt * 32 + half * 8);
  }
  f32x8 acc0, acc1;                          // dual independent WMMA chains
  for (int s = 0; s < S; ++s) {
    const int c  = s / KB;
    const int kb = s - c * KB;
#if HAS_TDM
    if (issuer) {
      if (s + 1 < S) {
        const int c1 = (s + 1) / KB, kb1 = (s + 1) - c1 * KB;
        tdm_load_slab(offNext, W + (size_t)(c1 * SLAB_ROWS) * Kfull + kb1 * 512,
                      (u32)Kfull);
        tdm_wait(1);                         // TDM in-order: stage s complete
      } else {
        tdm_wait(0);
      }
    }
    __syncthreads();                         // slab(s) visible; compute(s-1) retired
    const u16* sb = slabCur + bRow;
#else
    copy_slab(slab0, W + (size_t)(c * SLAB_ROWS) * Kfull + kb * 512, Kfull);
    __syncthreads();
    const u16* sb = slab0 + bRow;
#endif
    if (kb == 0) {
      const float bv = bias[c * 32 + jSel * 16 + ln16];
#pragma unroll
      for (int r = 0; r < 8; ++r) { acc0[r] = bv; acc1[r] = 0.0f; }
    }
    if constexpr (A_GLOBAL) {
      const u16* ap = aBase + kb * 512;
      bf16x16 a0 = load_a(ap + half * 8);
      bf16x16 p0 = load_b(sb + half * 8);
      bf16x16 a1 = load_a(ap + 32 + half * 8);
      bf16x16 p1 = load_b(sb + 32 + half * 8);
#pragma unroll
      for (int kt = 0; kt < 16; kt += 2) {
        const int k2 = (kt + 2) & 15, k3 = (kt + 3) & 15;  // wrap: dead loads at end
        acc0 = __builtin_amdgcn_wmma_f32_16x16x32_bf16(false, a0, false, p0, (short)0, acc0, false, false);
        a0 = load_a(ap + k2 * 32 + half * 8);
        p0 = load_b(sb + k2 * 32 + half * 8);
        acc1 = __builtin_amdgcn_wmma_f32_16x16x32_bf16(false, a1, false, p1, (short)0, acc1, false, false);
        a1 = load_a(ap + k3 * 32 + half * 8);
        p1 = load_b(sb + k3 * 32 + half * 8);
        SCHED_DS4_WMMA2();
      }
    } else {
      bf16x16 p0 = load_b(sb + half * 8);
      bf16x16 p1 = load_b(sb + 32 + half * 8);
#pragma unroll
      for (int kt = 0; kt < 16; kt += 2) {
        const int k2 = (kt + 2) & 15, k3 = (kt + 3) & 15;
        acc0 = __builtin_amdgcn_wmma_f32_16x16x32_bf16(false, Atile[kt], false, p0, (short)0, acc0, false, false);
        p0 = load_b(sb + k2 * 32 + half * 8);
        acc1 = __builtin_amdgcn_wmma_f32_16x16x32_bf16(false, Atile[kt + 1], false, p1, (short)0, acc1, false, false);
        p1 = load_b(sb + k3 * 32 + half * 8);
        SCHED_DS4_WMMA2();
      }
    }
    if (kb == KB - 1) {
      const int col = c * 32 + jSel * 16 + ln16;
#pragma unroll
      for (int r = 0; r < 8; ++r) {
        const int m = r + half * 8;          // C/D layout: VGPR r -> M = r + 8*half
        float v = acc0[r] + acc1[r];         // fold dual chains
        if (act == ACT_RELU_RES) v += bf2f(resW[m * LDS_STRIDE + col]);
        if (act != ACT_NONE)     v = fmaxf(v, 0.0f);
        dstW[m * LDS_STRIDE + col] = f2bf(v);
      }
    }
#if HAS_TDM
    slabCur = (slabCur == slab2) ? slab0 : (slabCur + SLAB_BYTES / 2);
    offNext = (offNext == SLAB2_OFF) ? SLAB0_OFF : (offNext + (u32)SLAB_BYTES);
#endif
  }
}

// Wave-local LayerNorm over 16 rows (2 lanes/row, shfl_xor(16)); low waves only.
__device__ void layernorm_wave(u16* waveBase, const float* __restrict__ g,
                               const float* __restrict__ b, int ln16, int half) {
  u16* p = waveBase + ln16 * LDS_STRIDE + half * 256;
  float s = 0.0f, sq = 0.0f;
  for (int i = 0; i < 256; i += 8) {
    const uint4 q = *(const uint4*)(p + i);
    const u32 w[4] = {q.x, q.y, q.z, q.w};
#pragma unroll
    for (int k = 0; k < 4; ++k) {
      const float lo = bf2f((u16)(w[k] & 0xffffu));
      const float hi = bf2f((u16)(w[k] >> 16));
      s += lo + hi; sq += lo * lo + hi * hi;
    }
  }
  s  += __shfl_xor(s, 16, 32);
  sq += __shfl_xor(sq, 16, 32);
  const float mean = s * (1.0f / 512.0f);
  const float var  = fmaxf(sq * (1.0f / 512.0f) - mean * mean, 0.0f);
  const float rstd = rsqrtf(var + 1e-5f);
  for (int i = 0; i < 256; i += 8) {
    const uint4 q = *(const uint4*)(p + i);
    const u32 w[4] = {q.x, q.y, q.z, q.w};
    uint4 o; u32* ow = (u32*)&o;
#pragma unroll
    for (int k = 0; k < 4; ++k) {
      const int c = half * 256 + i + k * 2;
      const float lo = (bf2f((u16)(w[k] & 0xffffu)) - mean) * rstd * g[c]     + b[c];
      const float hi = (bf2f((u16)(w[k] >> 16))     - mean) * rstd * g[c + 1] + b[c + 1];
      ow[k] = (u32)f2bf(lo) | ((u32)f2bf(hi) << 16);
    }
    *(uint4*)(p + i) = o;
  }
}

// Final 512 -> 1 linear + sigmoid (block3 Wout), f32 weights; low waves only.
__device__ void final_dot(const u16* __restrict__ srcW, const float* __restrict__ w1,
                          const float* __restrict__ b1, float* __restrict__ out,
                          int rowBase, int ln16, int half) {
  const u16* p = srcW + ln16 * LDS_STRIDE + half * 256;
  const float* w = w1 + half * 256;
  float s = 0.0f;
  for (int i = 0; i < 256; i += 8) {
    const uint4 q = *(const uint4*)(p + i);
    const u32 ww[4] = {q.x, q.y, q.z, q.w};
#pragma unroll
    for (int k = 0; k < 4; ++k) {
      s += bf2f((u16)(ww[k] & 0xffffu)) * w[i + k * 2];
      s += bf2f((u16)(ww[k] >> 16))     * w[i + k * 2 + 1];
    }
  }
  s += __shfl_xor(s, 16, 32);
  if (half == 0) {
    const float z = s + b1[0];
    out[rowBase + ln16] = 1.0f / (1.0f + __expf(-z));
  }
}

struct BlkP {
  const u16 *Wv, *Wav, *Wao, *Wd, *Wout;                 // bf16 (Wout null for blk3)
  const float *bv, *bav, *bao, *bd, *bout, *g, *b;       // f32
};
struct FusedP {
  const u16* xbf;
  float* out;
  const u16* Whead;
  const float* bhead;
  BlkP blk[3];
  const float* Wout3;   // f32 [1][512]
  const float* bout3;   // f32 [1]
};

__global__ __launch_bounds__(THREADS, 4) void fused_mlp(FusedP P) {
  extern __shared__ __align__(16) u16 smem[];
  u16* bufR  = smem;                                    // block input / residual
  u16* bufA  = bufR + ROWS_PER_WG * LDS_STRIDE;
  u16* bufB  = bufA + ROWS_PER_WG * LDS_STRIDE;
  u16* slab0 = smem + SLAB0_OFF / 2;
  u16* slab1 = smem + SLAB1_OFF / 2;
  u16* slab2 = smem + SLAB2_OFF / 2;

  const int lane    = threadIdx.x & 31;
  const int wave    = threadIdx.x >> 5;                 // 0..7
  const int ln16    = lane & 15;
  const int half    = lane >> 4;
  const int jSel    = wave >> 2;                        // N-tile within slab chunk
  const int waveRow = (wave & 3) * 16;                  // M-tile rows
  const int rowBase = blockIdx.x * ROWS_PER_WG + waveRow;
  const bool issuer = (wave == 0);                      // wave 0 issues TDM
  const bool lowW   = (jSel == 0);

  u16* R  = bufR + waveRow * LDS_STRIDE;
  u16* A  = bufA + waveRow * LDS_STRIDE;
  u16* Bb = bufB + waveRow * LDS_STRIDE;

  // head: h = relu(x @ head_W.T + head_b) -> R   (A from global bf16 x, K=2048)
  gemm_tiled<true, 4>(P.xbf + (size_t)(rowBase + ln16) * IN_DIM, R, nullptr,
                      P.Whead, P.bhead, slab0, slab1, slab2, ACT_RELU,
                      ln16, half, jSel, issuer);

  for (int ib = 0; ib < 3; ++ib) {
    const BlkP q = P.blk[ib];
    gemm_tiled<false, 1>(R,  A,  nullptr, q.Wv,  q.bv,  slab0, slab1, slab2, ACT_RELU, ln16, half, jSel, issuer);
    gemm_tiled<false, 1>(A,  Bb, nullptr, q.Wav, q.bav, slab0, slab1, slab2, ACT_NONE, ln16, half, jSel, issuer);
    gemm_tiled<false, 1>(Bb, A,  nullptr, q.Wao, q.bao, slab0, slab1, slab2, ACT_NONE, ln16, half, jSel, issuer);
    __syncthreads();                         // both pair waves' cols visible
    if (lowW) layernorm_wave(A, q.g, q.b, ln16, half);
    gemm_tiled<false, 1>(A,  Bb, nullptr, q.Wd,  q.bd,  slab0, slab1, slab2, ACT_RELU, ln16, half, jSel, issuer);
    if (ib < 2) {
      gemm_tiled<false, 1>(Bb, R, R, q.Wout, q.bout, slab0, slab1, slab2, ACT_RELU_RES, ln16, half, jSel, issuer);
    } else {
      __syncthreads();                       // both pair waves' cols visible
      if (lowW) final_dot(Bb, P.Wout3, P.bout3, P.out, rowBase, ln16, half);
    }
  }
}

// f32 -> bf16 conversion (4 elems/thread)
__global__ void cvt_f32_bf16(const float* __restrict__ src, u16* __restrict__ dst, long n4) {
  const long i = (long)blockIdx.x * blockDim.x + threadIdx.x;
  if (i >= n4) return;
  const float4 f = ((const float4*)src)[i];
  const u32 a = (u32)f2bf(f.x) | ((u32)f2bf(f.y) << 16);
  const u32 b = (u32)f2bf(f.z) | ((u32)f2bf(f.w) << 16);
  ((uint2*)dst)[i] = make_uint2(a, b);
}

// d_in order (setup_inputs insertion order):
// [0]=x, [1]=head_W, [2]=head_b,
// block i at base=3+12*i: +0 Wv, +1 bv, +2 attn_Wv, +3 attn_bv, +4 attn_Wo,
//   +5 attn_bo, +6 ln_g, +7 ln_b, +8 Wd, +9 bd, +10 Wout, +11 bout
extern "C" void kernel_launch(void* const* d_in, const int* in_sizes, int n_in,
                              void* d_out, int out_size, void* d_ws, size_t ws_size,
                              hipStream_t stream) {
  (void)in_sizes; (void)n_in; (void)out_size; (void)ws_size;
  u16* ws = (u16*)d_ws;
  size_t off = 0;

  auto cvt = [&](const void* s, u16* dptr, size_t n) {
    const long n4 = (long)(n / 4);
    const int thr = 256;
    const long blocks = (n4 + thr - 1) / thr;
    hipLaunchKernelGGL(cvt_f32_bf16, dim3((u32)blocks), dim3(thr), 0, stream,
                       (const float*)s, dptr, n4);
  };

  FusedP P{};
  u16* xbf = ws + off; off += (size_t)BATCH * IN_DIM;
  cvt(d_in[0], xbf, (size_t)BATCH * IN_DIM);
  u16* Whead = ws + off; off += (size_t)DIM * IN_DIM;
  cvt(d_in[1], Whead, (size_t)DIM * IN_DIM);
  P.xbf = xbf;
  P.out = (float*)d_out;
  P.Whead = Whead;
  P.bhead = (const float*)d_in[2];

  for (int ib = 0; ib < 3; ++ib) {
    const int base = 3 + 12 * ib;
    const int widx[5] = {0, 2, 4, 8, 10};
    const u16* Wc[5];
    for (int j = 0; j < 5; ++j) {
      if (ib == 2 && j == 4) { Wc[j] = nullptr; continue; }   // block3 Wout stays f32
      u16* d = ws + off; off += (size_t)DIM * DIM;
      cvt(d_in[base + widx[j]], d, (size_t)DIM * DIM);
      Wc[j] = d;
    }
    BlkP& q = P.blk[ib];
    q.Wv = Wc[0]; q.Wav = Wc[1]; q.Wao = Wc[2]; q.Wd = Wc[3]; q.Wout = Wc[4];
    q.bv   = (const float*)d_in[base + 1];
    q.bav  = (const float*)d_in[base + 3];
    q.bao  = (const float*)d_in[base + 5];
    q.g    = (const float*)d_in[base + 6];
    q.b    = (const float*)d_in[base + 7];
    q.bd   = (const float*)d_in[base + 9];
    q.bout = (const float*)d_in[base + 11];
  }
  P.Wout3 = (const float*)d_in[3 + 24 + 10];
  P.bout3 = (const float*)d_in[3 + 24 + 11];

  hipFuncSetAttribute((const void*)fused_mlp,
                      hipFuncAttributeMaxDynamicSharedMemorySize, SMEM_BYTES);
  hipLaunchKernelGGL(fused_mlp, dim3(BATCH / ROWS_PER_WG), dim3(THREADS),
                     SMEM_BYTES, stream, P);
}
